// GatedAttentionUnit_15805479649285
// MI455X (gfx1250) — matrix operand order repaired and checked by reading, not compiled, not run on hardware
//
#include <hip/hip_runtime.h>
#include <hip/hip_bf16.h>

typedef __attribute__((ext_vector_type(16))) _Float16 v16h;
typedef __attribute__((ext_vector_type(8)))  _Float16 v8h;
typedef __attribute__((ext_vector_type(8)))  float    v8f;

union U16h { v16h v; _Float16 h[16]; };
union U8h  { v8h  v; _Float16 h[8];  };
union U8f  { v8f  v; float    f[8];  };

#define HEAD_D 128
#define E_DIM  1536
#define SEQ    2048
#define BATCH  4
#define HID    768
#define NI     3200   // 2E + HEAD
#define ROWS   8192   // BATCH*SEQ

// ---------------------------------------------------------------------------
// fp32 -> fp16 convert (row-major copy)
// ---------------------------------------------------------------------------
__global__ void f32_to_f16_kernel(const float* __restrict__ in,
                                  _Float16* __restrict__ out, size_t n) {
  size_t i = (size_t)blockIdx.x * 256 + threadIdx.x;
  if (i < n) out[i] = (_Float16)in[i];
}

// ---------------------------------------------------------------------------
// fp32 [R x C] -> fp16 transposed [C x R], 32x32 LDS tile
// ---------------------------------------------------------------------------
__global__ __launch_bounds__(256) void transpose_f32_to_f16_kernel(
    const float* __restrict__ in, _Float16* __restrict__ out, int R, int C) {
  __shared__ float tile[32][33];
  int c0 = blockIdx.x * 32, r0 = blockIdx.y * 32;
  int tx = threadIdx.x & 31, ty = threadIdx.x >> 5;  // 32 x 8
#pragma unroll
  for (int i = 0; i < 32; i += 8)
    tile[ty + i][tx] = in[(size_t)(r0 + ty + i) * C + (c0 + tx)];
  __syncthreads();
#pragma unroll
  for (int i = 0; i < 32; i += 8)
    out[(size_t)(c0 + ty + i) * R + (r0 + tx)] = (_Float16)tile[tx][ty + i];
}

// ---------------------------------------------------------------------------
// WMMA GEMM: C = A(f16,[M x K] row-major) x Bt(f16,[N x ldB], K contiguous)
// 128x128 block tile, 8 waves (4x2), each wave 32x64 (2x4 tiles), K-step 32.
// All LDS traffic is b128; fragments are contiguous v8h pairs per ISA layout.
// MODE 0: +bias, SiLU, split-store u_t/v_t (f16,[E x ROWS]) + qk (f32)
// MODE 1: gate by u_t, store f16 g row-major       (P x V)
// MODE 2: +bias, store f32                          (final projection)
// ---------------------------------------------------------------------------
template <int MODE>
__global__ __launch_bounds__(256) void gemm_wmma_kernel(
    const _Float16* __restrict__ A, const _Float16* __restrict__ Bt,
    int M, int N, int K, int ldB,
    const float* __restrict__ bias,
    _Float16* __restrict__ outUt, _Float16* __restrict__ outVt,
    float* __restrict__ outQK,
    const _Float16* __restrict__ gateUt, _Float16* __restrict__ outG,
    float* __restrict__ outF,
    size_t batchStrideA, size_t batchOffB, int batchRowOff) {
  const int tid  = threadIdx.x;
  const int lane = tid & 31;
  const int wave = tid >> 5;
  const int wm   = wave & 3;     // 4 waves along M (32 rows each)
  const int wn   = wave >> 2;    // 2 waves along N (64 cols each)
  const int bm   = blockIdx.y * 128;
  const int bn   = blockIdx.x * 128;
  const int zb   = blockIdx.z;

  A  += (size_t)zb * batchStrideA;
  Bt += (size_t)zb * batchOffB;
  const int rowoff = zb * batchRowOff;

  __shared__ __align__(16) _Float16 As[128][32];
  __shared__ __align__(16) _Float16 Bs[128][32];   // [n][k]

  v8f acc[2][4] = {};

  const int mrow = lane & 15;          // A fragment row / B fragment col
  const int grp  = lane >> 4;          // half-wave select

  // per-thread staging coordinates (same for A and B tiles)
  const int sIdx0 = tid * 8;           // half index in 128x32 tile, round 0
  const int sRow0 = sIdx0 >> 5, sCol0 = sIdx0 & 31;
  const int sRow1 = sRow0 + 64;        // round 1 = +2048 halves = +64 rows

  // hoisted global staging pointers (stride 32 halves per K step)
  const _Float16* aP0 = A + (size_t)(bm + sRow0) * K + sCol0;
  const _Float16* aP1 = A + (size_t)(bm + sRow1) * K + sCol0;
  const _Float16* bP0 = Bt + (size_t)(bn + sRow0) * ldB + sCol0;
  const _Float16* bP1 = Bt + (size_t)(bn + sRow1) * ldB + sCol0;

  for (int k0 = 0; k0 < K; k0 += 32) {
    *(v8h*)&As[sRow0][sCol0] = *(const v8h*)aP0;
    *(v8h*)&As[sRow1][sCol0] = *(const v8h*)aP1;
    *(v8h*)&Bs[sRow0][sCol0] = *(const v8h*)bP0;
    *(v8h*)&Bs[sRow1][sCol0] = *(const v8h*)bP1;
    if (k0 + 32 < K) {  // prefetch next K tile (global_prefetch_b8)
      __builtin_prefetch(aP0 + 32, 0, 0);
      __builtin_prefetch(bP0 + 32, 0, 0);
    }
    aP0 += 32; aP1 += 32; bP0 += 32; bP1 += 32;
    __syncthreads();

    v16h afrag[2];
#pragma unroll
    for (int mi = 0; mi < 2; ++mi) {
      const v8h* ap = (const v8h*)&As[wm * 32 + mi * 16 + mrow][0];  // 4 chunks
      U16h af;
      *(v8h*)&af.h[0] = ap[grp];        // K = hk..hk+7
      *(v8h*)&af.h[8] = ap[2 + grp];    // K = 16+hk..16+hk+7
      afrag[mi] = af.v;
    }
#pragma unroll
    for (int ni = 0; ni < 4; ++ni) {
      const v8h* bp = (const v8h*)&Bs[wn * 64 + ni * 16 + mrow][0];
      U16h bf;
      *(v8h*)&bf.h[0] = bp[grp];           // K = hk..hk+7
      *(v8h*)&bf.h[8] = bp[2 + grp];       // K = 16+hk..16+hk+7
#pragma unroll
      for (int mi = 0; mi < 2; ++mi) {
        acc[mi][ni] = __builtin_amdgcn_wmma_f32_16x16x32_f16(
            false, afrag[mi], false, bf.v, (short)0, acc[mi][ni], false, false);
      }
    }
    __syncthreads();
  }

  // epilogue
#pragma unroll
  for (int mi = 0; mi < 2; ++mi) {
#pragma unroll
    for (int ni = 0; ni < 4; ++ni) {
      U8f c; c.v = acc[mi][ni];
      const int n     = bn + wn * 64 + ni * 16 + mrow;
      const int mbase = bm + wm * 32 + mi * 16 + grp * 8;  // 8 consecutive rows
      if (MODE == 0) {
        float bsv = bias[n];
        U8h pk;
#pragma unroll
        for (int r = 0; r < 8; ++r) {
          float val = c.f[r] + bsv;
          val = val / (1.0f + __expf(-val));  // SiLU
          c.f[r] = val;
          pk.h[r] = (_Float16)val;
        }
        if (n < E_DIM) {
          // u_t[e][m]: 8 consecutive m -> single b128 store
          *(v8h*)(outUt + (size_t)n * ROWS + mbase) = pk.v;
        } else if (n < 2 * E_DIM) {
          *(v8h*)(outVt + (size_t)(n - E_DIM) * ROWS + mbase) = pk.v;
        } else {
          float* qb = outQK + (size_t)mbase * HEAD_D + (n - 2 * E_DIM);
#pragma unroll
          for (int r = 0; r < 8; ++r) qb[r * HEAD_D] = c.f[r];
        }
      } else if (MODE == 1) {
        const size_t gm = (size_t)(rowoff + mbase);
        U8h ug; ug.v = *(const v8h*)(gateUt + (size_t)n * ROWS + gm);
        _Float16* gb = outG + gm * E_DIM + n;
#pragma unroll
        for (int r = 0; r < 8; ++r)
          gb[r * E_DIM] = (_Float16)(c.f[r] * (float)ug.h[r]);
      } else {
        float bsv = bias[n];
        float* ob = outF + (size_t)mbase * HID + n;
#pragma unroll
        for (int r = 0; r < 8; ++r) ob[r * HID] = c.f[r] + bsv;
      }
    }
  }
}

// ---------------------------------------------------------------------------
// OffsetScale + RoPE: qk(f32) -> q,k (f16, row-major [row,128])
// ---------------------------------------------------------------------------
__global__ __launch_bounds__(128) void rope_qk_kernel(
    const float* __restrict__ qk, const float* __restrict__ gamma,
    const float* __restrict__ beta, _Float16* __restrict__ qh,
    _Float16* __restrict__ kh) {
  int row = blockIdx.x;          // 0..8191
  int d   = threadIdx.x;         // 0..127
  __shared__ float sq[HEAD_D];
  sq[d] = qk[(size_t)row * HEAD_D + d];
  __syncthreads();
  int pos = row & (SEQ - 1);
  int i   = d >> 1;
  float freq = __expf(-((float)(2 * i) * (1.0f / 128.0f)) * 9.210340371976184f);
  float ang  = (float)pos * freq;
  float c = __cosf(ang), s = __sinf(ang);
  int   nb = (d & 1) ? (d - 1) : (d + 1);
  float x = sq[d], xn = sq[nb];
  float qv = x * gamma[d] + beta[d];
  float qn = xn * gamma[nb] + beta[nb];
  float q2 = (d & 1) ? qn : -qn;
  qh[(size_t)row * HEAD_D + d] = (_Float16)(qv * c + q2 * s);
  float kv = x * gamma[HEAD_D + d] + beta[HEAD_D + d];
  float kn = xn * gamma[HEAD_D + nb] + beta[HEAD_D + nb];
  float k2 = (d & 1) ? kn : -kn;
  kh[(size_t)row * HEAD_D + d] = (_Float16)(kv * c + k2 * s);
}

// ---------------------------------------------------------------------------
// softmax_plus length scale per batch: log(max(sum(mask),1))/log(512)
// ---------------------------------------------------------------------------
__global__ __launch_bounds__(256) void mask_scale_kernel(
    const float* __restrict__ mask, float* __restrict__ scaleB) {
  int b = blockIdx.x, tid = threadIdx.x;
  __shared__ float red[256];
  float s = 0.f;
  for (int j = tid; j < SEQ; j += 256) s += mask[(size_t)b * SEQ + j];
  red[tid] = s;
  __syncthreads();
  for (int off = 128; off > 0; off >>= 1) {
    if (tid < off) red[tid] += red[tid + off];
    __syncthreads();
  }
  if (tid == 0)
    scaleB[b] = __logf(fmaxf(red[0], 1.0f)) * (1.0f / 6.2383246250395075f);
}

// ---------------------------------------------------------------------------
// scores = (q kT / sqrt(d) + addmask) * scale ; row softmax ; store P (f16)
// One block = one batch x 16 query rows. 8 waves each own 256 key columns,
// scores held entirely in 16 v8f accumulators per wave (WMMA q.kT).
// All global accesses are base-pointer + compile-time immediate offsets.
// ---------------------------------------------------------------------------
__global__ __launch_bounds__(256) void scores_softmax_kernel(
    const _Float16* __restrict__ q, const _Float16* __restrict__ k,
    const float* __restrict__ mask, const float* __restrict__ scaleB,
    _Float16* __restrict__ P) {
  const int b    = blockIdx.y;
  const int qm   = blockIdx.x * 16;
  const int tid  = threadIdx.x;
  const int lane = tid & 31;
  const int w    = tid >> 5;
  const int mrow = lane & 15;
  const int grp  = lane >> 4;

  __shared__ float partMax[8][16], partSum[8][16];
  __shared__ float rowMax[16], rowSum[16];

  // hoisted base pointers: all subsequent offsets are unroll-constant
  const _Float16* qbase =
      q + ((size_t)b * SEQ + qm + mrow) * HEAD_D + grp * 8;
  const _Float16* kbase =
      k + ((size_t)b * SEQ + w * 256 + mrow) * HEAD_D + grp * 8;
  const float* mbase = mask + (size_t)b * SEQ + w * 256 + mrow;
  _Float16* pbase =
      P + ((size_t)b * SEQ + qm + grp * 8) * SEQ + w * 256 + mrow;

  // q fragments for the 16 rows (K = 128 -> 4 steps of 32)
  v16h aq[4];
#pragma unroll
  for (int ks = 0; ks < 4; ++ks) {
    U16h t;
    *(v8h*)&t.h[0] = *(const v8h*)(qbase + ks * 32);
    *(v8h*)&t.h[8] = *(const v8h*)(qbase + ks * 32 + 16);
    aq[ks] = t.v;
  }

  const float sb   = scaleB[b];
  const float invs = 0.08838834764831845f;  // 1/sqrt(128)

  v8f acc[16];
#pragma unroll
  for (int ct = 0; ct < 16; ++ct) {
    v8f a = {};
#pragma unroll
    for (int ks = 0; ks < 4; ++ks) {
      U16h bf;  // B column = k row (j0+mrow), K contiguous in d
      *(v8h*)&bf.h[0] = *(const v8h*)(kbase + ct * 16 * HEAD_D + ks * 32);
      *(v8h*)&bf.h[8] = *(const v8h*)(kbase + ct * 16 * HEAD_D + ks * 32 + 16);
      a = __builtin_amdgcn_wmma_f32_16x16x32_f16(false, aq[ks], false, bf.v,
                                                 (short)0, a, false, false);
    }
    float addm = (1.0f - mbase[ct * 16]) * -1e12f;
    U8f c; c.v = a;
#pragma unroll
    for (int r = 0; r < 8; ++r) c.f[r] = (c.f[r] * invs + addm) * sb;
    acc[ct] = c.v;
  }

  // row max (rows grp*8+r), reduce across the 16 N-lanes then across waves
  float mx[8];
#pragma unroll
  for (int r = 0; r < 8; ++r) {
    float m = -3.0e38f;
#pragma unroll
    for (int ct = 0; ct < 16; ++ct) { U8f c; c.v = acc[ct]; m = fmaxf(m, c.f[r]); }
    for (int off = 1; off < 16; off <<= 1) m = fmaxf(m, __shfl_xor(m, off, 32));
    mx[r] = m;
  }
  if (mrow == 0)
#pragma unroll
    for (int r = 0; r < 8; ++r) partMax[w][grp * 8 + r] = mx[r];
  __syncthreads();
  if (tid < 16) {
    float m = -3.0e38f;
    for (int ww = 0; ww < 8; ++ww) m = fmaxf(m, partMax[ww][tid]);
    rowMax[tid] = m;
  }
  __syncthreads();

  // exp and row sums (overwrite acc with exp values)
  float sm[8];
#pragma unroll
  for (int r = 0; r < 8; ++r) {
    float rm = rowMax[grp * 8 + r];
    float s = 0.f;
#pragma unroll
    for (int ct = 0; ct < 16; ++ct) {
      U8f c; c.v = acc[ct];
      float e = __expf(c.f[r] - rm);
      c.f[r] = e;
      acc[ct] = c.v;
      s += e;
    }
    for (int off = 1; off < 16; off <<= 1) s += __shfl_xor(s, off, 32);
    sm[r] = s;
  }
  if (mrow == 0)
#pragma unroll
    for (int r = 0; r < 8; ++r) partSum[w][grp * 8 + r] = sm[r];
  __syncthreads();
  if (tid < 16) {
    float s = 0.f;
    for (int ww = 0; ww < 8; ++ww) s += partSum[ww][tid];
    rowSum[tid] = s;
  }
  __syncthreads();

  // normalize and store P as f16 (immediate offsets from pbase)
#pragma unroll
  for (int ct = 0; ct < 16; ++ct) {
    U8f c; c.v = acc[ct];
#pragma unroll
    for (int r = 0; r < 8; ++r) {
      float p = c.f[r] / rowSum[grp * 8 + r];
      pbase[r * SEQ + ct * 16] = (_Float16)p;
    }
  }
}

// ---------------------------------------------------------------------------
extern "C" void kernel_launch(void* const* d_in, const int* in_sizes, int n_in,
                              void* d_out, int out_size, void* d_ws,
                              size_t ws_size, hipStream_t stream) {
  const float* hs    = (const float*)d_in[0];
  const float* mask  = (const float*)d_in[1];
  const float* Wi    = (const float*)d_in[2];
  const float* bi    = (const float*)d_in[3];
  const float* gamma = (const float*)d_in[4];
  const float* beta  = (const float*)d_in[5];
  const float* Wo    = (const float*)d_in[6];
  const float* bo    = (const float*)d_in[7];
  float* out = (float*)d_out;

  char* ws = (char*)d_ws;
  size_t off = 0;
  auto take = [&](size_t bytes) -> char* {
    char* p = ws + off;
    off = (off + bytes + 255) & ~(size_t)255;
    return p;
  };
  _Float16* hs_h = (_Float16*)take((size_t)ROWS * HID * 2);
  _Float16* Wi_t = (_Float16*)take((size_t)HID * NI * 2);     // [NI x HID]
  _Float16* Wo_t = (_Float16*)take((size_t)E_DIM * HID * 2);  // [HID x E]
  _Float16* u_t  = (_Float16*)take((size_t)ROWS * E_DIM * 2); // [E x ROWS]
  _Float16* v_t  = (_Float16*)take((size_t)ROWS * E_DIM * 2); // [E x ROWS]
  float*    qk_f = (float*)take((size_t)ROWS * HEAD_D * 4);
  _Float16* q_h  = (_Float16*)take((size_t)ROWS * HEAD_D * 2);
  _Float16* k_h  = (_Float16*)take((size_t)ROWS * HEAD_D * 2);
  _Float16* P    = (_Float16*)take((size_t)BATCH * SEQ * SEQ * 2);
  _Float16* g_h  = (_Float16*)take((size_t)ROWS * E_DIM * 2); // row-major
  float*    scB  = (float*)take(4 * sizeof(float));

  // converts
  {
    size_t n = (size_t)ROWS * HID;
    f32_to_f16_kernel<<<(n + 255) / 256, 256, 0, stream>>>(hs, hs_h, n);
  }
  transpose_f32_to_f16_kernel<<<dim3(NI / 32, HID / 32), 256, 0, stream>>>(
      Wi, Wi_t, HID, NI);
  transpose_f32_to_f16_kernel<<<dim3(HID / 32, E_DIM / 32), 256, 0, stream>>>(
      Wo, Wo_t, E_DIM, HID);

  // GEMM1: x = silu(hs @ Wi + bi) -> u_t(f16), v_t(f16), qk(f32)
  gemm_wmma_kernel<0><<<dim3(NI / 128, ROWS / 128, 1), 256, 0, stream>>>(
      hs_h, Wi_t, ROWS, NI, HID, HID, bi, u_t, v_t, qk_f, nullptr, nullptr,
      nullptr, 0, 0, 0);

  // OffsetScale + RoPE
  rope_qk_kernel<<<ROWS, 128, 0, stream>>>(qk_f, gamma, beta, q_h, k_h);

  // softmax_plus scale per batch
  mask_scale_kernel<<<BATCH, 256, 0, stream>>>(mask, scB);

  // scores + softmax -> P
  scores_softmax_kernel<<<dim3(SEQ / 16, BATCH), 256, 0, stream>>>(
      q_h, k_h, mask, scB, P);

  // g = u * (P @ v)   (per batch; v_t column offset = b*SEQ)
  gemm_wmma_kernel<1>
      <<<dim3(E_DIM / 128, SEQ / 128, BATCH), 256, 0, stream>>>(
          P, v_t, SEQ, E_DIM, SEQ, ROWS, nullptr, nullptr, nullptr, nullptr,
          u_t, g_h, nullptr, (size_t)SEQ * SEQ, (size_t)SEQ, SEQ);

  // out = g @ Wo + bo
  gemm_wmma_kernel<2><<<dim3(HID / 128, ROWS / 128, 1), 256, 0, stream>>>(
      g_h, Wo_t, ROWS, HID, E_DIM, E_DIM, bo, nullptr, nullptr, nullptr,
      nullptr, nullptr, out, 0, 0, 0);
}
